// Net_56246891708512
// MI455X (gfx1250) — 8-bit matrix operand order repaired and checked by reading, compiled, not run on hardware
//
#include <hip/hip_runtime.h>
#include <cstdint>

typedef int v8i __attribute__((ext_vector_type(8)));
typedef int v4i __attribute__((ext_vector_type(4)));
typedef int v2i __attribute__((ext_vector_type(2)));
typedef unsigned int v4u __attribute__((ext_vector_type(4)));

__global__ void prep_weights_kernel(const float* __restrict__ W,
                                    const float* __restrict__ bias,
                                    int N, int Npad, int Ktrue, int Kpad,
                                    signed char* __restrict__ Wq,
                                    float* __restrict__ cterm) {
    int n = blockIdx.x * blockDim.x + threadIdx.x;
    if (n >= Npad) return;
    if (n < N) {
        int s = 0;
        for (int k = 0; k < Ktrue; ++k) {
            signed char q = (W[(size_t)n * Ktrue + k] != 0.0f) ? 1 : 0;
            Wq[(size_t)n * Kpad + k] = q;
            s += q;
        }
        for (int k = Ktrue; k < Kpad; ++k) Wq[(size_t)n * Kpad + k] = 0;
        cterm[n] = (float)s + bias[n] - 0.5f * (float)Ktrue;
    } else {
        for (int k = 0; k < Kpad; ++k) Wq[(size_t)n * Kpad + k] = 0;
        cterm[n] = 0.0f;
    }
}

#if __has_builtin(__builtin_amdgcn_tensor_load_to_lds) && \
    __has_builtin(__builtin_amdgcn_s_wait_tensorcnt)
#define CONV_USE_TDM 1
__device__ __forceinline__ void tdm_load_image(const float* gptr,
                                               unsigned lds_addr) {
    unsigned long long ga = (unsigned long long)(uintptr_t)gptr;
    v4u g0 = {1u,
              lds_addr,
              (unsigned)(ga & 0xffffffffu),
              (unsigned)((ga >> 32) & 0x01ffffffu)
                  | (2u << 30)};
    v8i g1;
    g1[0] = 0x20000;
    g1[1] = (int)(3072u << 16);
    g1[2] = (int)(1u << 16);
    g1[3] = (int)(3072u << 16);
    g1[4] = 1;
    g1[5] = 3072;
    g1[6] = 0;
    g1[7] = 0;
    v4i gz = {0, 0, 0, 0};
#if __clang_major__ >= 23
    v8i gz8;
#pragma unroll
    for (int j = 0; j < 8; ++j) gz8[j] = 0;
    __builtin_amdgcn_tensor_load_to_lds(g0, g1, gz, gz, gz8, 0);
#else
    __builtin_amdgcn_tensor_load_to_lds(g0, g1, gz, gz, 0);
#endif
}
#endif

__global__ __launch_bounds__(256) void conv_bin_kernel(
    const float* __restrict__ x, const float* __restrict__ cw,
    const float* __restrict__ cb, signed char* __restrict__ Xq,
    float* __restrict__ rs0) {
    __shared__ float img[3072];
    __shared__ float w[80];
    __shared__ int   red[256];

    const int b = blockIdx.x;
    const int t = threadIdx.x;
    const float* xb = x + (size_t)b * 3072;

#ifdef CONV_USE_TDM
    if (t < 32) {
        tdm_load_image(xb, (unsigned)(uintptr_t)img);
        __builtin_amdgcn_s_wait_tensorcnt(0);
    }
#else
    for (int i = t; i < 768; i += 256)
        ((float4*)img)[i] = ((const float4*)xb)[i];
#endif
    if (t < 75) w[t] = cw[t];
    if (t == 75) w[75] = cb[0];
    __syncthreads();

    int cnt = 0;
    signed char* xq = Xq + (size_t)b * 832;
    for (int o = t; o < 784; o += 256) {
        const int oy = o / 28, ox = o % 28;
        float acc = w[75];
#pragma unroll
        for (int c = 0; c < 3; ++c)
#pragma unroll
            for (int ky = 0; ky < 5; ++ky)
#pragma unroll
                for (int kx = 0; kx < 5; ++kx)
                    acc += img[c * 1024 + (oy + ky) * 32 + (ox + kx)] *
                           w[c * 25 + ky * 5 + kx];
        signed char q = (acc != 0.0f) ? 1 : 0;
        xq[o] = q;
        cnt += q;
    }
    for (int o = 784 + t; o < 832; o += 256) xq[o] = 0;

    red[t] = cnt;
    __syncthreads();
    for (int s = 128; s > 0; s >>= 1) {
        if (t < s) red[t] += red[t + s];
        __syncthreads();
    }
    if (t == 0) rs0[b] = (float)red[0];
}

__global__ void rowsum_kernel(const unsigned char* __restrict__ H, int words,
                              float* __restrict__ rs, int rows) {
    int r = blockIdx.x * blockDim.x + threadIdx.x;
    if (r >= rows) return;
    const unsigned int* p = (const unsigned int*)(H + (size_t)r * (words * 4));
    unsigned int sum = 0;
    for (int w = 0; w < words; ++w) sum += (p[w] * 0x01010101u) >> 24;
    rs[r] = (float)sum;
}

__device__ __forceinline__ v8i load_a_chunk(const signed char* arow, int kb) {
    v2i t0 = *(const v2i*)(arow + kb);
    v2i t1 = *(const v2i*)(arow + kb + 8);
    v2i t2 = *(const v2i*)(arow + kb + 32);
    v2i t3 = *(const v2i*)(arow + kb + 40);
    v8i a;
    a[0] = t0[0]; a[1] = t0[1];
    a[2] = t1[0]; a[3] = t1[1];
    a[4] = t2[0]; a[5] = t2[1];
    a[6] = t3[0]; a[7] = t3[1];
    return a;
}

__device__ __forceinline__ v8i load_b_chunk(const signed char* wrow) {
    v4i b0 = *(const v4i*)(wrow);
    v4i b1 = *(const v4i*)(wrow + 32);
    v8i b;
    b[0] = b0[0]; b[1] = b0[1]; b[2] = b0[2]; b[3] = b0[3];
    b[4] = b1[0]; b[5] = b1[1]; b[6] = b1[2]; b[7] = b1[3];
    return b;
}

template <int N_TILES, int K_CHUNKS>
__global__ __launch_bounds__(256) void xor_gemm_bin(
    const signed char* __restrict__ Xq, int kpad,
    const signed char* __restrict__ Wq,
    const float* __restrict__ rowsum, const float* __restrict__ cterm,
    signed char* __restrict__ Hq, int nstride) {
    const int lane = threadIdx.x & 31;
    const int wid  = (blockIdx.x * blockDim.x + threadIdx.x) >> 5;
    const int m0   = wid * 16;
    const int half = lane >> 4;
    const int ln   = lane & 15;

    const signed char* arow = Xq + (size_t)(m0 + ln) * kpad;

    v8i acc[N_TILES];
#pragma unroll
    for (int nt = 0; nt < N_TILES; ++nt)
#pragma unroll
        for (int j = 0; j < 8; ++j) acc[nt][j] = 0;

#pragma unroll
    for (int c = 0; c < K_CHUNKS; ++c) {
        const int kb = c * 64 + half * 16;
        if (c + 1 < K_CHUNKS) __builtin_prefetch(arow + kb + 64, 0, 0);
        const v8i a = load_a_chunk(arow, kb);
#pragma unroll
        for (int nt = 0; nt < N_TILES; ++nt) {
            const signed char* wrow =
                Wq + (size_t)(nt * 16 + ln) * kpad + c * 64 + half * 16;
            const v8i b = load_b_chunk(wrow);
            acc[nt] = __builtin_amdgcn_wmma_i32_16x16x64_iu8(
                false, a, false, b, acc[nt], false, false);
        }
    }

    float rs[8];
#pragma unroll
    for (int v = 0; v < 8; ++v) rs[v] = rowsum[m0 + half * 8 + v];
#pragma unroll
    for (int nt = 0; nt < N_TILES; ++nt) {
        const int n = nt * 16 + ln;
        const float ct = cterm[n];
#pragma unroll
        for (int v = 0; v < 8; ++v) {
            float s = rs[v] + ct - 2.0f * (float)acc[nt][v];
            Hq[(size_t)(m0 + half * 8 + v) * nstride + n] =
                (s >= 0.0f) ? (signed char)1 : (signed char)0;
        }
    }
}

__global__ __launch_bounds__(256) void xor_gemm_final(
    const signed char* __restrict__ Xq,
    const signed char* __restrict__ Wq,
    const float* __restrict__ rowsum, const float* __restrict__ cterm,
    float* __restrict__ out) {
    const int lane = threadIdx.x & 31;
    const int wid  = (blockIdx.x * blockDim.x + threadIdx.x) >> 5;
    const int m0   = wid * 16;
    const int half = lane >> 4;
    const int ln   = lane & 15;

    const signed char* arow = Xq + (size_t)(m0 + ln) * 64;
    const v8i a = load_a_chunk(arow, half * 16);
    const v8i b = load_b_chunk(Wq + (size_t)ln * 64 + half * 16);

    v8i acc;
#pragma unroll
    for (int j = 0; j < 8; ++j) acc[j] = 0;
    acc = __builtin_amdgcn_wmma_i32_16x16x64_iu8(false, a, false, b, acc,
                                                 false, false);

    const bool valid = (ln < 10);
    const float ct = cterm[ln];
#pragma unroll
    for (int v = 0; v < 8; ++v) {
        const int m = m0 + half * 8 + v;
        float s = rowsum[m] + ct - 2.0f * (float)acc[v];
        float logit = valid ? s : -1e30f;
        float mx = logit;
#pragma unroll
        for (int mask = 1; mask < 16; mask <<= 1)
            mx = fmaxf(mx, __shfl_xor(mx, mask, 16));
        float e = valid ? __expf(s - mx) : 0.0f;
        float sm = e;
#pragma unroll
        for (int mask = 1; mask < 16; mask <<= 1)
            sm += __shfl_xor(sm, mask, 16);
        if (valid) out[(size_t)m * 10 + ln] = s - mx - logf(sm);
    }
}

extern "C" void kernel_launch(void* const* d_in, const int* in_sizes, int n_in,
                              void* d_out, int out_size, void* d_ws,
                              size_t ws_size, hipStream_t stream) {
    const float* x  = (const float*)d_in[0];
    const float* cw = (const float*)d_in[1];
    const float* cb = (const float*)d_in[2];
    const float* W1 = (const float*)d_in[3];
    const float* b1 = (const float*)d_in[4];
    const float* W2 = (const float*)d_in[5];
    const float* b2 = (const float*)d_in[6];
    const float* W3 = (const float*)d_in[7];
    const float* b3 = (const float*)d_in[8];
    float* out = (float*)d_out;

    const int B = in_sizes[0] / 3072;

    char* ws = (char*)d_ws;
    size_t off = 0;
    auto take = [&](size_t bytes) -> char* {
        char* p = ws + off;
        off += (bytes + 255) & ~(size_t)255;
        return p;
    };
    signed char* Xq  = (signed char*)take((size_t)B * 832);
    signed char* h1  = (signed char*)take((size_t)B * 128);
    signed char* h2  = (signed char*)take((size_t)B * 64);
    float* rs0 = (float*)take((size_t)B * 4);
    float* rs1 = (float*)take((size_t)B * 4);
    float* rs2 = (float*)take((size_t)B * 4);
    signed char* Wq1 = (signed char*)take(128 * 832);
    signed char* Wq2 = (signed char*)take(64 * 128);
    signed char* Wq3 = (signed char*)take(16 * 64);
    float* ct1 = (float*)take(128 * 4);
    float* ct2 = (float*)take(64 * 4);
    float* ct3 = (float*)take(16 * 4);

    prep_weights_kernel<<<2, 64, 0, stream>>>(W1, b1, 128, 128, 784, 832, Wq1, ct1);
    prep_weights_kernel<<<1, 64, 0, stream>>>(W2, b2, 64, 64, 128, 128, Wq2, ct2);
    prep_weights_kernel<<<1, 16, 0, stream>>>(W3, b3, 10, 16, 64, 64, Wq3, ct3);

    conv_bin_kernel<<<B, 256, 0, stream>>>(x, cw, cb, Xq, rs0);

    const int waves  = B / 16;
    const int blocks = waves * 32 / 256;

    xor_gemm_bin<8, 13><<<blocks, 256, 0, stream>>>(Xq, 832, Wq1, rs0, ct1, h1, 128);
    rowsum_kernel<<<B / 256, 256, 0, stream>>>((const unsigned char*)h1, 32, rs1, B);

    xor_gemm_bin<4, 2><<<blocks, 256, 0, stream>>>(h1, 128, Wq2, rs1, ct2, h2, 64);
    rowsum_kernel<<<B / 256, 256, 0, stream>>>((const unsigned char*)h2, 16, rs2, B);

    xor_gemm_final<<<blocks, 256, 0, stream>>>(h2, Wq3, rs2, ct3, out);
}
